// GenePerturbationAttention_79250736545844
// MI455X (gfx1250) — matrix operand order repaired and checked by reading, not compiled, not run on hardware
//
#include <hip/hip_runtime.h>

// ---------------------------------------------------------------------------
// GenePerturbationAttention for MI455X (gfx1250, wave32, WMMA f16 -> f32 acc)
//   B=256, G=20000, LATENT=512, H=8, HD=64
//   out = [gene_attention (B*G fp32) | attn (B*H*G fp32)]
// Round 2: 32x32 register tile per wave (2x2 WMMA frags) -> 1.0 fragment
// loads per v_wmma instead of 2.0; block = 8 waves covering 64 x 128.
// ---------------------------------------------------------------------------

#define B_   256
#define G_   20000
#define L_   512
#define GD_  256
#define H_   8
#define HD_  64
#define SCALE_ 0.125f        // HD^-0.5
#define ATTN_UPSCALE 4096.0f // keep attn probs (~5e-5) in f16 normal range

typedef __attribute__((ext_vector_type(16))) _Float16 v16h;
typedef __attribute__((ext_vector_type(8)))  _Float16 v8h;
typedef __attribute__((ext_vector_type(8)))  float    v8f;

// ---------------------------------------------------------------------------
// fp32 -> fp16 conversion (grid-stride)
// ---------------------------------------------------------------------------
__global__ void __launch_bounds__(256)
f32_to_f16_kernel(const float* __restrict__ src, _Float16* __restrict__ dst, size_t n) {
    size_t i = (size_t)blockIdx.x * blockDim.x + threadIdx.x;
    size_t stride = (size_t)gridDim.x * blockDim.x;
    for (; i < n; i += stride) dst[i] = (_Float16)src[i];
}

// ---------------------------------------------------------------------------
// Fragment loaders (wave32; A and B share this code's contiguous k order, not the ISA table order):
//   A 16x32 f16: m = lane&15; element e -> k = e + 16*(lane>>4)
//   B 32x16 f16: n = lane&15; element e -> k = e + 16*(lane>>4)
//   C/D 16x16 f32: vgpr r -> m = r + 8*(lane>>4), n = lane&15
// ---------------------------------------------------------------------------
__device__ __forceinline__ v16h load_a_frag(const _Float16* __restrict__ arow, int k0, int hi) {
    const v8h lo = *(const v8h*)(arow + k0 + 16 * hi);
    const v8h hv = *(const v8h*)(arow + k0 + 16 * hi + 8);
    v16h a;
#pragma unroll
    for (int e = 0; e < 8; ++e) { a[e] = lo[e]; a[e + 8] = hv[e]; }
    return a;
}

// ---------------------------------------------------------------------------
// Generic WMMA GEMM:  C[m,n] = scale * ( sum_k A[m,k]*B'[k,n] + bias[n] )
//   B_IS_WEIGHT=true : B given as W[N,K] row-major (B' = W^T), contiguous frag
//   B_IS_WEIGHT=false: B given as [K,N] row-major
// 32x32 C tile per wave (4 accumulators), 8 waves/block = 64 rows x 128 cols.
// Requires M, N multiples of 32 (all shapes here qualify); partial blocks
// guarded at wave granularity. grid.z batches heads via element strides.
// ---------------------------------------------------------------------------
template <bool B_IS_WEIGHT, bool OUT_HALF>
__global__ void __launch_bounds__(256)
gemm_wmma_kernel(const _Float16* __restrict__ A, int lda, long aBatch,
                 const _Float16* __restrict__ Bm, int ldb, long bBatch,
                 void* __restrict__ Cv, int ldc, long cBatch,
                 const float* __restrict__ bias, float scale,
                 int M, int N, int K)
{
    const int z = blockIdx.z;
    A  += (size_t)z * (size_t)aBatch;
    Bm += (size_t)z * (size_t)bBatch;

    const int tid  = threadIdx.x;
    const int wave = tid >> 5;
    const int lane = tid & 31;
    const int ln   = lane & 15;
    const int hi   = lane >> 4;

    const int m0 = blockIdx.y * 64 + (wave >> 2) * 32;   // 32-row wave tile
    const int n0 = blockIdx.x * 128 + (wave & 3) * 32;   // 32-col wave tile
    if (m0 >= M || n0 >= N) return;                      // M,N are multiples of 32

    const _Float16* __restrict__ arow0 = A + (size_t)(m0 + ln)      * (size_t)lda;
    const _Float16* __restrict__ arow1 = A + (size_t)(m0 + 16 + ln) * (size_t)lda;
    const _Float16* __restrict__ brow0 = Bm + (B_IS_WEIGHT ? (size_t)(n0 + ln)      * (size_t)ldb : 0);
    const _Float16* __restrict__ brow1 = Bm + (B_IS_WEIGHT ? (size_t)(n0 + 16 + ln) * (size_t)ldb : 0);

    v8f c00 = {}, c01 = {}, c10 = {}, c11 = {};
    for (int k0 = 0; k0 < K; k0 += 32) {
        const v16h a0 = load_a_frag(arow0, k0, hi);
        const v16h a1 = load_a_frag(arow1, k0, hi);
        v16h b0, b1;
        if (B_IS_WEIGHT) {
            b0 = *(const v16h*)(brow0 + k0 + 16 * hi);   // contiguous 32B
            b1 = *(const v16h*)(brow1 + k0 + 16 * hi);
        } else {
#pragma unroll
            for (int e = 0; e < 16; ++e) {
                const size_t r = (size_t)(k0 + e + 16 * hi) * (size_t)ldb;
                b0[e] = Bm[r + (n0 + ln)];
                b1[e] = Bm[r + (n0 + 16 + ln)];
            }
        }
        c00 = __builtin_amdgcn_wmma_f32_16x16x32_f16(false, a0, false, b0, (short)0, c00, false, false);
        c01 = __builtin_amdgcn_wmma_f32_16x16x32_f16(false, a0, false, b1, (short)0, c01, false, false);
        c10 = __builtin_amdgcn_wmma_f32_16x16x32_f16(false, a1, false, b0, (short)0, c10, false, false);
        c11 = __builtin_amdgcn_wmma_f32_16x16x32_f16(false, a1, false, b1, (short)0, c11, false, false);
    }

    const float bn0 = bias ? bias[n0 + ln]      : 0.0f;
    const float bn1 = bias ? bias[n0 + 16 + ln] : 0.0f;
    const size_t cz = (size_t)z * (size_t)cBatch;
#pragma unroll
    for (int r = 0; r < 8; ++r) {
        const int mA = m0 + r + 8 * hi;
        const int mB = mA + 16;
        const size_t iA0 = cz + (size_t)mA * (size_t)ldc + (n0 + ln);
        const size_t iA1 = iA0 + 16;
        const size_t iB0 = cz + (size_t)mB * (size_t)ldc + (n0 + ln);
        const size_t iB1 = iB0 + 16;
        const float v00 = (c00[r] + bn0) * scale;
        const float v01 = (c01[r] + bn1) * scale;
        const float v10 = (c10[r] + bn0) * scale;
        const float v11 = (c11[r] + bn1) * scale;
        if (OUT_HALF) {
            _Float16* Ch = (_Float16*)Cv;
            Ch[iA0] = (_Float16)v00; Ch[iA1] = (_Float16)v01;
            Ch[iB0] = (_Float16)v10; Ch[iB1] = (_Float16)v11;
        } else {
            float* Cf = (float*)Cv;
            Cf[iA0] = v00; Cf[iA1] = v01;
            Cf[iB0] = v10; Cf[iB1] = v11;
        }
    }
}

// ---------------------------------------------------------------------------
// Row softmax over G elems, in place on fp32, plus upscaled f16 copy.
// One block (256 threads) per (b,h) row.
// ---------------------------------------------------------------------------
__global__ void __launch_bounds__(256)
softmax_kernel(float* __restrict__ s, _Float16* __restrict__ sh) {
    __shared__ float red[256];
    const int tid = threadIdx.x;
    const size_t base = (size_t)blockIdx.x * G_;

    float m = -3.4e38f;
    for (int i = tid; i < G_; i += 256) m = fmaxf(m, s[base + i]);
    red[tid] = m; __syncthreads();
    for (int t = 128; t > 0; t >>= 1) {
        if (tid < t) red[tid] = fmaxf(red[tid], red[tid + t]);
        __syncthreads();
    }
    m = red[0]; __syncthreads();

    float sum = 0.f;
    for (int i = tid; i < G_; i += 256) {
        float e = __expf(s[base + i] - m);
        s[base + i] = e;
        sum += e;
    }
    red[tid] = sum; __syncthreads();
    for (int t = 128; t > 0; t >>= 1) {
        if (tid < t) red[tid] += red[tid + t];
        __syncthreads();
    }
    const float inv = 1.0f / red[0];

    for (int i = tid; i < G_; i += 256) {
        float p = s[base + i] * inv;
        s[base + i]  = p;                              // exact fp32 attn output
        sh[base + i] = (_Float16)(p * ATTN_UPSCALE);   // f16 copy for ctx GEMM
    }
}

// ---------------------------------------------------------------------------
// LayerNorm(512) + ReLU, one block per batch row; thread t owns elems t, t+256
// ---------------------------------------------------------------------------
__global__ void __launch_bounds__(256)
ln_relu_kernel(const float* __restrict__ hin, const float* __restrict__ g,
               const float* __restrict__ bta, _Float16* __restrict__ hout) {
    __shared__ float red[256];
    const int tid = threadIdx.x;
    const size_t base = (size_t)blockIdx.x * L_;

    const float x0 = hin[base + tid];
    const float x1 = hin[base + tid + 256];

    red[tid] = x0 + x1; __syncthreads();
    for (int t = 128; t > 0; t >>= 1) { if (tid < t) red[tid] += red[tid + t]; __syncthreads(); }
    const float mean = red[0] * (1.0f / L_); __syncthreads();

    const float d0 = x0 - mean, d1 = x1 - mean;
    red[tid] = d0 * d0 + d1 * d1; __syncthreads();
    for (int t = 128; t > 0; t >>= 1) { if (tid < t) red[tid] += red[tid + t]; __syncthreads(); }
    const float inv = rsqrtf(red[0] * (1.0f / L_) + 1e-5f);

    const float y0 = fmaxf(d0 * inv * g[tid]       + bta[tid],       0.f);
    const float y1 = fmaxf(d1 * inv * g[tid + 256] + bta[tid + 256], 0.f);
    hout[base + tid]       = (_Float16)y0;
    hout[base + tid + 256] = (_Float16)y1;
}

// ---------------------------------------------------------------------------
// Host-side pipeline
// ---------------------------------------------------------------------------
extern "C" void kernel_launch(void* const* d_in, const int* in_sizes, int n_in,
                              void* d_out, int out_size, void* d_ws, size_t ws_size,
                              hipStream_t stream) {
    (void)in_sizes; (void)n_in; (void)out_size; (void)ws_size;

    const float* z   = (const float*)d_in[0];   // [B,L]
    const float* ge  = (const float*)d_in[1];   // [G,GD]
    const float* Wq  = (const float*)d_in[2];   // [L,L]
    const float* bq  = (const float*)d_in[3];
    const float* Wk  = (const float*)d_in[4];   // [L,GD]
    const float* bk  = (const float*)d_in[5];
    const float* Wv  = (const float*)d_in[6];   // [L,GD]
    const float* bv  = (const float*)d_in[7];
    const float* W1  = (const float*)d_in[8];   // [L,L]
    const float* b1  = (const float*)d_in[9];
    const float* lng = (const float*)d_in[10];
    const float* lnb = (const float*)d_in[11];
    const float* W2  = (const float*)d_in[12];  // [G,L]
    const float* b2  = (const float*)d_in[13];

    float* out_gene = (float*)d_out;                       // [B,G]
    float* out_attn = out_gene + (size_t)B_ * G_;          // [B,H,G]

    // ---- workspace carve-out (256B aligned slices; total ~157 MB)
    char* wp = (char*)d_ws;
    auto carve = [&](size_t bytes) -> char* {
        char* p = wp;
        wp += (bytes + 255) & ~(size_t)255;
        return p;
    };
    _Float16* ge_h  = (_Float16*)carve((size_t)G_ * GD_ * 2);
    _Float16* Wk_h  = (_Float16*)carve((size_t)L_ * GD_ * 2);
    _Float16* Wv_h  = (_Float16*)carve((size_t)L_ * GD_ * 2);
    _Float16* Wq_h  = (_Float16*)carve((size_t)L_ * L_ * 2);
    _Float16* W1_h  = (_Float16*)carve((size_t)L_ * L_ * 2);
    _Float16* W2_h  = (_Float16*)carve((size_t)G_ * L_ * 2);
    _Float16* z_h   = (_Float16*)carve((size_t)B_ * L_ * 2);
    _Float16* K_h   = (_Float16*)carve((size_t)G_ * L_ * 2);
    _Float16* V_h   = (_Float16*)carve((size_t)G_ * L_ * 2);
    _Float16* Q_h   = (_Float16*)carve((size_t)B_ * L_ * 2);
    _Float16* at_h  = (_Float16*)carve((size_t)B_ * H_ * G_ * 2);
    _Float16* ctx_h = (_Float16*)carve((size_t)B_ * L_ * 2);
    float*    h_f   = (float*)   carve((size_t)B_ * L_ * 4);
    _Float16* h2_h  = (_Float16*)carve((size_t)B_ * L_ * 2);

    // ---- fp32 -> fp16 conversions
    f32_to_f16_kernel<<<1024, 256, 0, stream>>>(ge, ge_h, (size_t)G_ * GD_);
    f32_to_f16_kernel<<<256,  256, 0, stream>>>(Wk, Wk_h, (size_t)L_ * GD_);
    f32_to_f16_kernel<<<256,  256, 0, stream>>>(Wv, Wv_h, (size_t)L_ * GD_);
    f32_to_f16_kernel<<<256,  256, 0, stream>>>(Wq, Wq_h, (size_t)L_ * L_);
    f32_to_f16_kernel<<<256,  256, 0, stream>>>(W1, W1_h, (size_t)L_ * L_);
    f32_to_f16_kernel<<<2048, 256, 0, stream>>>(W2, W2_h, (size_t)G_ * L_);
    f32_to_f16_kernel<<<256,  256, 0, stream>>>(z,  z_h,  (size_t)B_ * L_);

    // ---- K = ge @ Wk^T + bk   [G,L] (f16)   M=20000 N=512 K=256
    gemm_wmma_kernel<true, true><<<dim3(L_ / 128, (G_ + 63) / 64, 1), 256, 0, stream>>>(
        ge_h, GD_, 0, Wk_h, GD_, 0, (void*)K_h, L_, 0, bk, 1.0f, G_, L_, GD_);
    // ---- V = ge @ Wv^T + bv   [G,L] (f16)
    gemm_wmma_kernel<true, true><<<dim3(L_ / 128, (G_ + 63) / 64, 1), 256, 0, stream>>>(
        ge_h, GD_, 0, Wv_h, GD_, 0, (void*)V_h, L_, 0, bv, 1.0f, G_, L_, GD_);
    // ---- Q = (z @ Wq^T + bq) * SCALE   [B,L] (f16), softmax scale folded in
    gemm_wmma_kernel<true, true><<<dim3(L_ / 128, B_ / 64, 1), 256, 0, stream>>>(
        z_h, L_, 0, Wq_h, L_, 0, (void*)Q_h, L_, 0, bq, SCALE_, B_, L_, L_);

    // ---- scores[b,h,g] = Q_head . K_head (fp32, pre-softmax, into attn out)
    //      per-head GEMM: A = Q (+h*64, lda=L), W = K (+h*64, ldw=L)
    //      C[(b*H+h)*G+g]: ldc=H*G, batch off = G
    gemm_wmma_kernel<true, false><<<dim3((G_ + 127) / 128, B_ / 64, H_), 256, 0, stream>>>(
        Q_h, L_, HD_, K_h, L_, HD_, (void*)out_attn, H_ * G_, G_,
        nullptr, 1.0f, B_, G_, HD_);

    // ---- softmax over G, in place; also f16 copy scaled by 4096
    softmax_kernel<<<B_ * H_, 256, 0, stream>>>(out_attn, at_h);

    // ---- ctx[b, h*64+d] = sum_g attn * V  (K=20000), 1/4096 folded in
    gemm_wmma_kernel<false, true><<<dim3(1, B_ / 64, H_), 256, 0, stream>>>(
        at_h, H_ * G_, G_, V_h, L_, HD_, (void*)ctx_h, L_, HD_,
        nullptr, 1.0f / ATTN_UPSCALE, B_, HD_, G_);

    // ---- h = ctx @ W1^T + b1  (fp32)
    gemm_wmma_kernel<true, false><<<dim3(L_ / 128, B_ / 64, 1), 256, 0, stream>>>(
        ctx_h, L_, 0, W1_h, L_, 0, (void*)h_f, L_, 0, b1, 1.0f, B_, L_, L_);

    // ---- LayerNorm + ReLU -> f16
    ln_relu_kernel<<<B_, 256, 0, stream>>>(h_f, lng, lnb, h2_h);

    // ---- gene_attention = h2 @ W2^T + b2  [B,G] fp32
    gemm_wmma_kernel<true, false><<<dim3((G_ + 127) / 128, B_ / 64, 1), 256, 0, stream>>>(
        h2_h, L_, 0, W2_h, L_, 0, (void*)out_gene, G_, 0, b2, 1.0f, B_, G_, L_);
}
